// GraphMETNetwork_57732950392992
// MI455X (gfx1250) — matrix operand order repaired and checked by reading, not compiled, not run on hardware
//
#include <hip/hip_runtime.h>
#include <hip/hip_bf16.h>

#define BB   32
#define NN   2048
#define HH   64
#define KK   20
#define NTOT (BB * NN)
#define NTILES (NN / 16)   // 128

typedef __attribute__((ext_vector_type(16))) _Float16 v16h;
typedef __attribute__((ext_vector_type(8)))  _Float16 v8h;
typedef __attribute__((ext_vector_type(8)))  float    v8f;

// async-to-LDS builtin operand types (per hipcc diagnostic: int4, AS1 / AS3)
typedef int v4i_ __attribute__((vector_size(16)));
typedef __attribute__((address_space(1))) v4i_ GAS4;
typedef __attribute__((address_space(3))) v4i_ LAS4;

__device__ __forceinline__ float elu1(float x) {
    return x > 0.0f ? x : (__expf(x) - 1.0f);
}

__device__ __forceinline__ v8f wmma_f16(v16h a, v16h b, v8f c) {
    return __builtin_amdgcn_wmma_f32_16x16x32_f16(
        /*neg_a=*/false, a, /*neg_b=*/false, b,
        /*c_mod=*/(short)0, c, /*reuse_a=*/false, /*reuse_b=*/false);
}

// A-fragment for V_WMMA_F32_16X16X32_F16 (this code's contiguous K order, same as B):
// lane (hi = lane>>4, m = lane&15), K-step `step` covers K = step*32 .. +31:
//   halfs [0..7]  = K = step*32 + hi*16 + 0..7
//   halfs [8..15] = K = step*32 + hi*16 + 8 + 0..7
__device__ __forceinline__ v16h ldA(const _Float16* row, int hi, int step) {
    const v8h x0 = *(const v8h*)(row + step * 32 + hi * 16);
    const v8h x1 = *(const v8h*)(row + step * 32 + hi * 16 + 8);
    v16h r;
#pragma unroll
    for (int i = 0; i < 8; ++i) { r[i] = x0[i]; r[i + 8] = x1[i]; }
    return r;
}

// ---------------------------------------------------------------------------
// 1) Fused node embedding: cont MLP + categorical embeddings + encoder (+ELU)
// ---------------------------------------------------------------------------
__global__ __launch_bounds__(256) void embed_kernel(
    const float* __restrict__ xc, const int* __restrict__ xcat,
    const float* __restrict__ Ech, const float* __restrict__ Epdg, const float* __restrict__ Epv,
    const float* __restrict__ Wc, const float* __restrict__ bc,
    const float* __restrict__ Wcat, const float* __restrict__ bcat,
    const float* __restrict__ Wenc, const float* __restrict__ benc,
    float* __restrict__ emb)
{
    int i = blockIdx.x * blockDim.x + threadIdx.x;
    if (i >= NTOT) return;

    float xin[8];
#pragma unroll
    for (int c = 0; c < 8; ++c) xin[c] = xc[i * 8 + c];

    float econt[32];
#pragma unroll 4
    for (int o = 0; o < 32; ++o) {
        float s = bc[o];
#pragma unroll
        for (int c = 0; c < 8; ++c) s += xin[c] * Wc[c * 32 + o];
        econt[o] = elu1(s);
    }

    int p = xcat[i * 3 + 0]; p = p < 0 ? -p : p;
    const int PD[7] = {1, 2, 11, 13, 22, 130, 211};
    int pi = p;
#pragma unroll
    for (int v = 0; v < 7; ++v) if (p == PD[v]) pi = v;
    int ch = xcat[i * 3 + 1] + 1;
    int pv = xcat[i * 3 + 2];

    float cat48[48];
#pragma unroll
    for (int c = 0; c < 16; ++c) {
        cat48[c]      = Ech[ch * 16 + c];
        cat48[16 + c] = Epdg[pi * 16 + c];
        cat48[32 + c] = Epv[pv * 16 + c];
    }

    float ecat[32];
#pragma unroll 2
    for (int o = 0; o < 32; ++o) {
        float s = bcat[o];
#pragma unroll 8
        for (int c = 0; c < 48; ++c) s += cat48[c] * Wcat[c * 32 + o];
        ecat[o] = elu1(s);
    }

    // encoder: input = [ecat(32), econt(32)]
#pragma unroll 2
    for (int o = 0; o < HH; ++o) {
        float s = benc[o];
#pragma unroll 8
        for (int c = 0; c < 32; ++c) s += ecat[c] * Wenc[c * HH + o];
#pragma unroll 8
        for (int c = 0; c < 32; ++c) s += econt[c] * Wenc[(32 + c) * HH + o];
        emb[(size_t)i * HH + o] = elu1(s);
    }
}

// ---------------------------------------------------------------------------
// 2) BatchNorm over all NTOT rows: stats via atomics, then (add-)apply
// ---------------------------------------------------------------------------
__global__ __launch_bounds__(256) void bn_stats_kernel(
    const float* __restrict__ x, float* __restrict__ stats)
{
    long idx    = (long)blockIdx.x * blockDim.x + threadIdx.x;
    long stride = (long)gridDim.x * blockDim.x;   // multiple of 64
    int  chn    = (int)(idx & 63);
    float s = 0.0f, ss = 0.0f;
    for (long e = idx; e < (long)NTOT * HH; e += stride) {
        float v = x[e];
        s += v; ss += v * v;
    }
    atomicAdd(&stats[chn], s);
    atomicAdd(&stats[64 + chn], ss);
}

__global__ __launch_bounds__(256) void bn_apply_kernel(
    float* __restrict__ emb, const float* __restrict__ src,
    const float* __restrict__ stats,
    const float* __restrict__ g, const float* __restrict__ b, int add_mode)
{
    long e = (long)blockIdx.x * blockDim.x + threadIdx.x;
    if (e >= (long)NTOT * HH) return;
    int chn = (int)(e & 63);
    const float inv_n = 1.0f / (float)NTOT;
    float m = stats[chn] * inv_n;
    float v = stats[64 + chn] * inv_n - m * m;
    float y = g[chn] * (src[e] - m) * rsqrtf(v + 1e-5f) + b[chn];
    if (add_mode) emb[e] += y; else emb[e] = y;
}

// ---------------------------------------------------------------------------
// 3) prep: emb(f32) -> Xh(f16) + squared row norms; convw: Wm(f32) -> f16
// ---------------------------------------------------------------------------
__global__ __launch_bounds__(256) void prep_kernel(
    const float* __restrict__ emb, _Float16* __restrict__ Xh, float* __restrict__ sqn)
{
    int i = blockIdx.x * blockDim.x + threadIdx.x;
    if (i >= NTOT) return;
    float s = 0.0f;
#pragma unroll 8
    for (int c = 0; c < HH; ++c) {
        float v = emb[(size_t)i * HH + c];
        s += v * v;
        Xh[(size_t)i * HH + c] = (_Float16)v;
    }
    sqn[i] = s;
}

__global__ __launch_bounds__(256) void convw_kernel(
    const float* __restrict__ Wm, _Float16* __restrict__ Wmh_g)
{
    int t = blockIdx.x * blockDim.x + threadIdx.x;
    if (t < 2 * 128 * HH) Wmh_g[t] = (_Float16)Wm[t];
}

// ---------------------------------------------------------------------------
// 4) KNN: 9 waves. Waves 0-7: WMMA gram stripes (double-buffered distance
//    stripe in LDS). Wave 8: streaming top-20 scan of the previous stripe,
//    overlapped with the current stripe's WMMA.
// ---------------------------------------------------------------------------
__global__ __launch_bounds__(288) void knn_kernel(
    const _Float16* __restrict__ Xh, const float* __restrict__ sqn,
    int* __restrict__ knn)
{
    __shared__ float Ds[2][16][136];   // double-buffered 128-col stripe (+pad)
    __shared__ float sqrow[16];

    int blk  = blockIdx.x;
    int b    = blk >> 7;               // / NTILES
    int tile = blk & (NTILES - 1);
    int rb   = tile * 16;

    int tid  = threadIdx.x;
    int lane = tid & 31;
    int wave = tid >> 5;               // 0..7 compute, 8 = scan
    int hi   = lane >> 4;
    int lm   = lane & 15;

    const _Float16* Xb = Xh + (size_t)b * NN * HH;

    if (tid < 16) sqrow[tid] = sqn[b * NN + rb + tid];
    __syncthreads();

    v16h a0{}, a1{};
    if (wave < 8) {
        const _Float16* arow = Xb + (size_t)(rb + lm) * HH;
        a0 = ldA(arow, hi, 0);
        a1 = ldA(arow, hi, 1);
    }

    float kd[KK]; int ki[KK];
#pragma unroll
    for (int k = 0; k < KK; ++k) { kd[k] = 3.0e38f; ki[k] = 0; }

    const int scan_row = tid - 256;    // 0..15 valid for scan wave

    for (int st = 0; st < NN / 128; ++st) {
        if (wave < 8) {
            int cb = st * 128 + wave * 16;
            const _Float16* crow = Xb + (size_t)(cb + lm) * HH;
            __builtin_prefetch(crow + 128 * HH, 0, 1);   // next stripe

            // B fragment: column n = lm, K = hi*16 .. +15 contiguous
            v16h b0 = *(const v16h*)(crow + hi * 16);
            v16h b1 = *(const v16h*)(crow + 32 + hi * 16);

            v8f c = {};
            c = wmma_f16(a0, b0, c);
            c = wmma_f16(a1, b1, c);

            float sqc = sqn[b * NN + cb + lm];
#pragma unroll
            for (int r = 0; r < 8; ++r) {
                int M = r + 8 * hi;
                Ds[st & 1][M][wave * 16 + lm] = sqrow[M] + sqc - 2.0f * c[r];
            }
        } else if (scan_row < 16 && st > 0) {
            const int pb = (st - 1) & 1;
            const int base = (st - 1) * 128;
            for (int col = 0; col < 128; ++col) {
                float d = Ds[pb][scan_row][col];
                if (d < kd[KK - 1]) {
                    int pos = KK - 1;
                    while (pos > 0 && kd[pos - 1] > d) {
                        kd[pos] = kd[pos - 1]; ki[pos] = ki[pos - 1]; --pos;
                    }
                    kd[pos] = d; ki[pos] = base + col;
                }
            }
        }
        __syncthreads();
    }

    // last stripe + writeout (scan wave only)
    if (wave == 8 && scan_row < 16) {
        const int pb = (NN / 128 - 1) & 1;
        const int base = NN - 128;
        for (int col = 0; col < 128; ++col) {
            float d = Ds[pb][scan_row][col];
            if (d < kd[KK - 1]) {
                int pos = KK - 1;
                while (pos > 0 && kd[pos - 1] > d) {
                    kd[pos] = kd[pos - 1]; ki[pos] = ki[pos - 1]; --pos;
                }
                kd[pos] = d; ki[pos] = base + col;
            }
        }
        int gi = b * NN + rb + scan_row;
#pragma unroll
        for (int k = 0; k < KK; ++k) knn[(size_t)gi * KK + k] = ki[k]; // in-batch idx
    }
}

// ---------------------------------------------------------------------------
// 5) Messages: per block = 16 nodes, 4 waves cover N=64. Wm f16 tile pulled
//    into LDS with GLOBAL_LOAD_ASYNC_TO_LDS_B128 (ASYNCcnt) when available.
//    For each of K=20 neighbors: build m_in = [xi, xj-xi] (16x128 f16) in
//    LDS, 4 WMMA K-steps against register-resident Wm fragments, running max.
// ---------------------------------------------------------------------------
__global__ __launch_bounds__(128) void msg_kernel(
    const _Float16* __restrict__ Xh, const int* __restrict__ knn,
    const _Float16* __restrict__ Wmh_g, const float* __restrict__ bm,
    float* __restrict__ agg, int layer)
{
    __shared__ _Float16 Wmh[128 * HH];   // 16 KB
    __shared__ _Float16 Min[16 * 128];   // 4 KB
    __shared__ int sj[16];

    const _Float16* Wg = Wmh_g + (size_t)layer * 128 * HH;

#if __has_builtin(__builtin_amdgcn_global_load_async_to_lds_b128)
    // async DMA of the 16KB weight tile into LDS (tracked by ASYNCcnt)
    for (int t = threadIdx.x; t < (128 * HH) / 8; t += 128) {
        char* g = (char*)Wg + (size_t)t * 16;     // const cast: source only read
        char* s = (char*)Wmh + (size_t)t * 16;
        __builtin_amdgcn_global_load_async_to_lds_b128((GAS4*)g, (LAS4*)s, 0, 0);
    }
#if __has_builtin(__builtin_amdgcn_s_wait_asynccnt)
    __builtin_amdgcn_s_wait_asynccnt(0);
#else
    asm volatile("s_wait_asynccnt 0x0" ::: "memory");
#endif
#else
    for (int t = threadIdx.x; t < 128 * HH; t += 128) Wmh[t] = Wg[t];
#endif
    __syncthreads();

    int blk  = blockIdx.x;
    int b    = blk >> 7;
    int tile = blk & (NTILES - 1);
    int nb   = tile * 16;

    int lane = threadIdx.x & 31;
    int wave = threadIdx.x >> 5;   // 0..3 -> N-tile
    int hi   = lane >> 4;
    int lm   = lane & 15;
    int nw   = wave * 16;

    // B fragments: B[k][n] = Wm[k][n]; col n = nw+lm, K = s*32 + hi*16 .. +15
    v16h bf[4];
#pragma unroll
    for (int s = 0; s < 4; ++s) {
#pragma unroll
        for (int q = 0; q < 16; ++q)
            bf[s][q] = Wmh[(s * 32 + hi * 16 + q) * HH + nw + lm];
    }

    const _Float16* Xb = Xh + (size_t)b * NN * HH;

    v8f rmax;
#pragma unroll
    for (int r = 0; r < 8; ++r) rmax[r] = -3.0e38f;

    for (int k = 0; k < KK; ++k) {
        if (threadIdx.x < 16)
            sj[threadIdx.x] = knn[(size_t)(b * NN + nb + threadIdx.x) * KK + k];
        __syncthreads();

        // build m_in tile: thread -> (row r, 16-half segment seg)
        {
            int r   = threadIdx.x >> 3;
            int seg = threadIdx.x & 7;
            const _Float16* xi = Xb + (size_t)(nb + r) * HH;
            const _Float16* xj = Xb + (size_t)sj[r] * HH;
#pragma unroll
            for (int q = 0; q < 16; ++q) {
                int c = seg * 16 + q;
                _Float16 v = (c < HH)
                    ? xi[c]
                    : (_Float16)((float)xj[c - HH] - (float)xi[c - HH]);
                Min[r * 128 + c] = v;
            }
        }
        __syncthreads();

        v8f c = {};
#pragma unroll
        for (int s = 0; s < 4; ++s) {
            v16h a = ldA(&Min[lm * 128], hi, s);
            c = wmma_f16(a, bf[s], c);
        }
#pragma unroll
        for (int r = 0; r < 8; ++r) rmax[r] = fmaxf(rmax[r], c[r]);
        __syncthreads();
    }

    // C layout: M = r + 8*hi (node in tile), N = nw + lm (channel)
#pragma unroll
    for (int r = 0; r < 8; ++r) {
        int node = b * NN + nb + r + 8 * hi;
        agg[(size_t)node * HH + nw + lm] = rmax[r] + bm[layer * HH + nw + lm];
    }
}

// ---------------------------------------------------------------------------
// 6) Output head: elu(emb @ Wo1 + bo1) @ Wo2 + bo2
// ---------------------------------------------------------------------------
__global__ __launch_bounds__(256) void head_kernel(
    const float* __restrict__ emb,
    const float* __restrict__ Wo1, const float* __restrict__ bo1,
    const float* __restrict__ Wo2, const float* __restrict__ bo2,
    float* __restrict__ out)
{
    int i = blockIdx.x * blockDim.x + threadIdx.x;
    if (i >= NTOT) return;
    float x[HH];
#pragma unroll 8
    for (int c = 0; c < HH; ++c) x[c] = emb[(size_t)i * HH + c];
    float s2 = bo2[0];
#pragma unroll 2
    for (int o = 0; o < 32; ++o) {
        float s = bo1[o];
#pragma unroll 8
        for (int c = 0; c < HH; ++c) s += x[c] * Wo1[c * 32 + o];
        s2 += elu1(s) * Wo2[o];
    }
    out[i] = s2;
}

// ---------------------------------------------------------------------------
extern "C" void kernel_launch(void* const* d_in, const int* in_sizes, int n_in,
                              void* d_out, int out_size, void* d_ws, size_t ws_size,
                              hipStream_t stream) {
    const float* x_cont = (const float*)d_in[0];
    const int*   x_cat  = (const int*)d_in[1];
    // d_in[2] edge_index, d_in[3] batch: unused by the reference forward
    const float* E_ch   = (const float*)d_in[4];
    const float* E_pdg  = (const float*)d_in[5];
    const float* E_pv   = (const float*)d_in[6];
    const float* Wc     = (const float*)d_in[7];
    const float* bc     = (const float*)d_in[8];
    const float* Wcat   = (const float*)d_in[9];
    const float* bcat   = (const float*)d_in[10];
    const float* Wenc   = (const float*)d_in[11];
    const float* benc   = (const float*)d_in[12];
    const float* g_all  = (const float*)d_in[13];
    const float* be_all = (const float*)d_in[14];
    const float* Wm     = (const float*)d_in[15];
    const float* bm     = (const float*)d_in[16];
    const float* g_bn   = (const float*)d_in[17];
    const float* b_bn   = (const float*)d_in[18];
    const float* Wo1    = (const float*)d_in[19];
    const float* bo1    = (const float*)d_in[20];
    const float* Wo2    = (const float*)d_in[21];
    const float* bo2    = (const float*)d_in[22];
    float* out = (float*)d_out;

    // workspace layout (256B aligned)
    char* ws = (char*)d_ws;
    size_t off = 0;
    auto alloc = [&](size_t bytes) {
        char* p = ws + off;
        off = (off + bytes + 255) & ~(size_t)255;
        return p;
    };
    float*    emb   = (float*)   alloc((size_t)NTOT * HH * sizeof(float));
    _Float16* Xh    = (_Float16*)alloc((size_t)NTOT * HH * sizeof(_Float16));
    float*    sqn   = (float*)   alloc((size_t)NTOT * sizeof(float));
    int*      knn   = (int*)     alloc((size_t)NTOT * KK * sizeof(int));
    float*    agg   = (float*)   alloc((size_t)NTOT * HH * sizeof(float));
    _Float16* Wmh_g = (_Float16*)alloc((size_t)2 * 128 * HH * sizeof(_Float16));
    float*    stats = (float*)   alloc(128 * sizeof(float));
    (void)ws_size; (void)in_sizes; (void)n_in; (void)out_size;

    const int elems = NTOT * HH;

    // 1) embed + encoder (pre-BN emb); f16 copy of message weights
    embed_kernel<<<NTOT / 256, 256, 0, stream>>>(
        x_cont, x_cat, E_ch, E_pdg, E_pv, Wc, bc, Wcat, bcat, Wenc, benc, emb);
    convw_kernel<<<(2 * 128 * HH) / 256, 256, 0, stream>>>(Wm, Wmh_g);

    // 2) global BN on emb
    (void)hipMemsetAsync(stats, 0, 128 * sizeof(float), stream);
    bn_stats_kernel<<<512, 256, 0, stream>>>(emb, stats);
    bn_apply_kernel<<<elems / 256, 256, 0, stream>>>(emb, emb, stats, g_all, be_all, 0);

    // 3) two EdgeConv-style layers
    for (int l = 0; l < 2; ++l) {
        prep_kernel<<<NTOT / 256, 256, 0, stream>>>(emb, Xh, sqn);
        knn_kernel<<<BB * NTILES, 288, 0, stream>>>(Xh, sqn, knn);
        msg_kernel<<<BB * NTILES, 128, 0, stream>>>(Xh, knn, Wmh_g, bm, agg, l);

        (void)hipMemsetAsync(stats, 0, 128 * sizeof(float), stream);
        bn_stats_kernel<<<512, 256, 0, stream>>>(agg, stats);
        bn_apply_kernel<<<elems / 256, 256, 0, stream>>>(
            emb, agg, stats, g_bn + l * HH, b_bn + l * HH, 1);
    }

    // 4) head
    head_kernel<<<NTOT / 256, 256, 0, stream>>>(emb, Wo1, bo1, Wo2, bo2, out);
}
